// TransformerBlock_66949950210215
// MI455X (gfx1250) — matrix operand order repaired and checked by reading, not compiled, not run on hardware
//
#include <hip/hip_runtime.h>

// ---------------------------------------------------------------------------
// TransformerBlock for MI455X (gfx1250): bf16 WMMA everywhere, flash attention,
// async global->LDS staging (GLOBAL_LOAD_ASYNC_TO_LDS_B128) with double buffer.
// Shapes: B=4, S=1024, E=512, H=8, HE=4096, R=B*S=4096, F1=4*E=2048.
// ---------------------------------------------------------------------------

typedef __attribute__((ext_vector_type(16))) __bf16          v16bf;
typedef __attribute__((ext_vector_type(8)))  float           v8f;
typedef __attribute__((ext_vector_type(8)))  unsigned short  v8u;
typedef __attribute__((ext_vector_type(16))) unsigned short  v16u;

__device__ __forceinline__ unsigned short f2bf(float f) {
  unsigned int u = __float_as_uint(f);
  u += 0x7FFFu + ((u >> 16) & 1u);        // round-to-nearest-even
  return (unsigned short)(u >> 16);
}

__device__ __forceinline__ v16bf make_frag(v8u lo, v8u hi) {
  v16u t = __builtin_shufflevector(lo, hi, 0,1,2,3,4,5,6,7,8,9,10,11,12,13,14,15);
  return __builtin_bit_cast(v16bf, t);
}

// A fragment (16x32 bf16): lane row = lane&15; lane half selects K {0..7,16..23} vs {8..15,24..31}
__device__ __forceinline__ v16bf ld_a(const unsigned short* base, int stride, int koff, int lane) {
  const unsigned short* p = base + (lane & 15) * stride + koff + ((lane >> 4) * 8);
  v8u lo = *(const v8u*)p;          // K = kb .. kb+7
  v8u hi = *(const v8u*)(p + 16);   // K = 16+kb .. 16+kb+7
  return make_frag(lo, hi);
}

// B fragment (32x16 bf16) from B^T storage [N][K]: lane col = lane&15; lane half selects K {0..7,16..23} vs {8..15,24..31}
__device__ __forceinline__ v16bf ld_bt(const unsigned short* base, size_t stride, int lane) {
  const unsigned short* p = base + (size_t)(lane & 15) * stride + ((lane >> 4) * 8);
  v8u lo = *(const v8u*)p;
  v8u hi = *(const v8u*)(p + 16);
  return make_frag(lo, hi);
}

__device__ __forceinline__ v8f wmma_bf16(v16bf a, v16bf b, v8f c) {
  return __builtin_amdgcn_wmma_f32_16x16x32_bf16(false, a, false, b, (short)0, c, false, false);
}

// ---- CDNA5 async global->LDS copy (ASYNCcnt-tracked DMA, no VGPR round trip)
// LDS byte address = low 32 bits of the generic address (ISA 10.2 aperture rule).
__device__ __forceinline__ unsigned lds_addr_of(const void* p) {
  return (unsigned)(uintptr_t)p;
}
__device__ __forceinline__ void async_b128(unsigned dst_lds, const void* src) {
  asm volatile("global_load_async_to_lds_b128 %0, %1, off"
               :: "v"(dst_lds), "v"(src)
               : "memory");
}
__device__ __forceinline__ void wait_async0() {
#if __has_builtin(__builtin_amdgcn_s_wait_asynccnt)
  __builtin_amdgcn_s_wait_asynccnt(0);
#else
  asm volatile("s_wait_asynccnt 0x0" ::: "memory");
#endif
}

// ---------------------------------------------------------------------------
// Elementwise f32 -> bf16
__global__ __launch_bounds__(256)
void f32_to_bf16_kernel(const float* __restrict__ in, unsigned short* __restrict__ out, int n) {
  int i = blockIdx.x * 256 + threadIdx.x;
  if (i < n) out[i] = f2bf(in[i]);
}

// Transpose-convert: W[K][N] f32 -> WT[N][K] bf16 (K, N multiples of 32)
__global__ __launch_bounds__(256)
void transpose_to_bf16_kernel(const float* __restrict__ W, unsigned short* __restrict__ WT,
                              int K, int N) {
  __shared__ float t[32][33];
  int kb = blockIdx.y * 32, nb = blockIdx.x * 32;
  int tx = threadIdx.x & 31, ty = threadIdx.x >> 5;   // ty in 0..7
  #pragma unroll
  for (int i = 0; i < 32; i += 8)
    t[ty + i][tx] = W[(size_t)(kb + ty + i) * N + nb + tx];
  __syncthreads();
  #pragma unroll
  for (int i = 0; i < 32; i += 8)
    WT[(size_t)(nb + ty + i) * K + kb + tx] = f2bf(t[tx][ty + i]);
}

// ---------------------------------------------------------------------------
// Generic bf16 WMMA GEMM: C[M,N] = A[M,K] @ B[K,N], B given transposed (BT[N][K]).
// Block tile 64x256, 8 waves (2x4), wave tile 32x64, K step 32, double-buffered
// async LDS staging (one barrier per K step, DMA overlapped with WMMA).
// mode 0: store bf16 row-major; mode 1: store f32 row-major; mode 2: bf16
// "V-transposed" store out[((row/S)*HE + col)*S + row%S].
__global__ __launch_bounds__(256)
void gemm_bf16_kernel(const unsigned short* __restrict__ A,
                      const unsigned short* __restrict__ BT,
                      const float* __restrict__ bias,
                      const float* __restrict__ resid,
                      unsigned short* __restrict__ outb,
                      float* __restrict__ outf,
                      int M, int N, int K, int mode, int relu,
                      int S_dim, int HE_dim) {
  __shared__ unsigned short sA[2][64 * 40];
  __shared__ unsigned short sB[2][256 * 40];
  const int tid  = threadIdx.x;
  const int lane = tid & 31, wave = tid >> 5;
  const int waveM = wave >> 2, waveN = wave & 3;
  const int bm = blockIdx.y * 64, bn = blockIdx.x * 256;

  // per-thread staging assignments (uniform across iterations)
  const int ar = tid >> 2, as = (tid & 3) * 8;       // A: 64 rows x 32, 16B each
  const unsigned short* aSrcBase = A + (size_t)(bm + ar) * K + as;
  const unsigned short* bSrcBase = BT + (size_t)(bn + tid) * K;

  auto stage = [&](int buf, int k0) {
    async_b128(lds_addr_of(&sA[buf][ar * 40 + as]), aSrcBase + k0);
    unsigned bdst = lds_addr_of(&sB[buf][tid * 40]);
    const unsigned short* bsrc = bSrcBase + k0;
    async_b128(bdst +  0, bsrc +  0);
    async_b128(bdst + 16, bsrc +  8);
    async_b128(bdst + 32, bsrc + 16);
    async_b128(bdst + 48, bsrc + 24);
  };

  v8f acc[2][4] = {};

  stage(0, 0);
  wait_async0();
  __syncthreads();

  int buf = 0;
  for (int k0 = 0; k0 < K; k0 += 32) {
    if (k0 + 32 < K) stage(buf ^ 1, k0 + 32);   // prefetch next tile (async DMA)

    v16bf af0 = ld_a(sA[buf] + (waveM * 32 +  0) * 40, 40, 0, lane);
    v16bf af1 = ld_a(sA[buf] + (waveM * 32 + 16) * 40, 40, 0, lane);
    #pragma unroll
    for (int ni = 0; ni < 4; ++ni) {
      v16bf bf = ld_bt(sB[buf] + (waveN * 64 + ni * 16) * 40, 40, lane);
      acc[0][ni] = wmma_bf16(af0, bf, acc[0][ni]);
      acc[1][ni] = wmma_bf16(af1, bf, acc[1][ni]);
    }

    wait_async0();        // our prefetch writes landed
    __syncthreads();      // everyone's reads (dscnt) + writes done
    buf ^= 1;
  }

  const int col16 = lane & 15, rbase = (lane >> 4) * 8;
  for (int mi = 0; mi < 2; ++mi) {
    for (int ni = 0; ni < 4; ++ni) {
      int col = bn + waveN * 64 + ni * 16 + col16;
      float bv = bias ? bias[col] : 0.f;
      #pragma unroll
      for (int r = 0; r < 8; ++r) {
        int row = bm + waveM * 32 + mi * 16 + rbase + r;
        float v = acc[mi][ni][r] + bv;
        if (resid) v += resid[(size_t)row * N + col];
        if (relu)  v = fmaxf(v, 0.f);
        if (mode == 0) {
          outb[(size_t)row * N + col] = f2bf(v);
        } else if (mode == 1) {
          outf[(size_t)row * N + col] = v;
        } else { // V-transposed: [B][HE][S]
          int b_ = row / S_dim, s_ = row - b_ * S_dim;
          outb[((size_t)b_ * HE_dim + col) * S_dim + s_] = f2bf(v);
        }
      }
    }
  }
}

// ---------------------------------------------------------------------------
// Flash attention: one workgroup = (b, h, 16-row Q tile). 8 waves.
// Scores: each wave computes its 16-key column slice over full d=E.
// PV: each wave owns a 64-wide slice of the E output columns.
// q,k layout [B,S,HE] bf16; vT layout [B,HE,S] bf16; out [B,S,HE] bf16.
__global__ __launch_bounds__(256)
void flash_attn_kernel(const unsigned short* __restrict__ q,
                       const unsigned short* __restrict__ k,
                       const unsigned short* __restrict__ vT,
                       unsigned short* __restrict__ out,
                       int S, int E, int Hh, float scale) {
  const int HE = Hh * E;
  const int qtiles = S / 16;
  int wg = blockIdx.x;
  int qt = wg % qtiles;
  int h  = (wg / qtiles) % Hh;
  int bb = wg / (qtiles * Hh);
  int q0 = qt * 16;

  const int tid = threadIdx.x, lane = tid & 31, wave = tid >> 5;
  const int col16 = lane & 15, rbase = (lane >> 4) * 8;

  __shared__ unsigned short Qs[16 * 520];   // Q tile 16 x E (padded)
  __shared__ float          sc[16 * 136];   // score tile 16 x 128 (padded)
  __shared__ unsigned short Ps[16 * 136];   // softmax'd probs, bf16
  __shared__ float m_sh[16], l_sh[16], rs_sh[16];

  { // stage Q tile once via async DMA (reused across all key tiles)
    int row = tid & 15, chunk = tid >> 4;   // 16 chunks of 32 elements
    const unsigned short* src = q + ((size_t)(bb * S + q0 + row) * HE) + h * E + chunk * 32;
    unsigned dst = lds_addr_of(Qs + row * 520 + chunk * 32);
    async_b128(dst +  0, src +  0);
    async_b128(dst + 16, src +  8);
    async_b128(dst + 32, src + 16);
    async_b128(dst + 48, src + 24);
  }
  if (tid < 16) { m_sh[tid] = -3.0e38f; l_sh[tid] = 0.f; rs_sh[tid] = 1.f; }
  wait_async0();
  __syncthreads();

  v8f acc[4] = {};

  for (int kt = 0; kt < S; kt += 128) {
    // ---- scores for this wave's 16 keys: QK^T over full d = E
    v8f s_acc = {};
    const unsigned short* kb_ptr = k + ((size_t)(bb * S + kt + wave * 16) * HE) + h * E;
    for (int kk = 0; kk < E; kk += 32) {
      v16bf a = ld_a(Qs, 520, kk, lane);
      v16bf b = ld_bt(kb_ptr + kk, (size_t)HE, lane);
      s_acc = wmma_bf16(a, b, s_acc);
    }
    #pragma unroll
    for (int r = 0; r < 8; ++r)
      sc[(rbase + r) * 136 + wave * 16 + col16] = s_acc[r] * scale;
    __syncthreads();

    // ---- online softmax stats (one thread per Q row)
    if (tid < 16) {
      int row = tid;
      float mold = m_sh[row], mmax = mold;
      for (int j = 0; j < 128; ++j) mmax = fmaxf(mmax, sc[row * 136 + j]);
      float lsum = 0.f;
      for (int j = 0; j < 128; ++j) {
        float p = __expf(sc[row * 136 + j] - mmax);
        Ps[row * 136 + j] = f2bf(p);
        lsum += p;
      }
      float rs = __expf(mold - mmax);
      l_sh[row] = l_sh[row] * rs + lsum;
      m_sh[row] = mmax;
      rs_sh[row] = rs;
    }
    __syncthreads();

    // ---- rescale accumulators, then P @ V for this wave's 64-col d slice
    #pragma unroll
    for (int nt = 0; nt < 4; ++nt)
      #pragma unroll
      for (int r = 0; r < 8; ++r)
        acc[nt][r] *= rs_sh[rbase + r];

    const unsigned short* vbase = vT + ((size_t)(bb * HE + h * E + wave * 64)) * S + kt;
    for (int kk2 = 0; kk2 < 128; kk2 += 32) {
      v16bf a = ld_a(Ps, 136, kk2, lane);
      #pragma unroll
      for (int nt = 0; nt < 4; ++nt) {
        v16bf b = ld_bt(vbase + (size_t)(nt * 16) * S + kk2, (size_t)S, lane);
        acc[nt] = wmma_bf16(a, b, acc[nt]);
      }
    }
    __syncthreads();
  }

  // ---- epilogue: divide by l, store bf16 [B,S,HE]
  for (int nt = 0; nt < 4; ++nt) {
    int col = h * E + wave * 64 + nt * 16 + col16;
    #pragma unroll
    for (int r = 0; r < 8; ++r) {
      int row = q0 + rbase + r;
      float v = acc[nt][r] / l_sh[rbase + r];
      out[(size_t)(bb * S + row) * HE + col] = f2bf(v);
    }
  }
}

// ---------------------------------------------------------------------------
// LayerNorm over N=512 (one block per row, 256 threads, 2 elems/thread).
__global__ __launch_bounds__(256)
void layernorm_kernel(const float* __restrict__ y, const float* __restrict__ g,
                      const float* __restrict__ bta, float* __restrict__ of,
                      unsigned short* __restrict__ ob, int N) {
  __shared__ float s1[256], s2[256];
  int row = blockIdx.x, tid = threadIdx.x;
  const float* yr = y + (size_t)row * N;
  float a = yr[tid], b = yr[tid + 256];
  s1[tid] = a + b;
  s2[tid] = a * a + b * b;
  __syncthreads();
  for (int off = 128; off > 0; off >>= 1) {
    if (tid < off) { s1[tid] += s1[tid + off]; s2[tid] += s2[tid + off]; }
    __syncthreads();
  }
  float mean = s1[0] / (float)N;
  float var  = s2[0] / (float)N - mean * mean;
  float rstd = rsqrtf(var + 1e-5f);
  float v1 = (a - mean) * rstd * g[tid]       + bta[tid];
  float v2 = (b - mean) * rstd * g[tid + 256] + bta[tid + 256];
  if (of) { of[(size_t)row * N + tid] = v1; of[(size_t)row * N + tid + 256] = v2; }
  if (ob) { ob[(size_t)row * N + tid] = f2bf(v1); ob[(size_t)row * N + tid + 256] = f2bf(v2); }
}

// ---------------------------------------------------------------------------
extern "C" void kernel_launch(void* const* d_in, const int* in_sizes, int n_in,
                              void* d_out, int out_size, void* d_ws, size_t ws_size,
                              hipStream_t stream) {
  const int Bb = 4, S = 1024, E = 512, H = 8;
  const int HE = H * E;        // 4096
  const int R  = Bb * S;       // 4096 token rows
  const int F1 = 4 * E;        // 2048

  const float* x  = (const float*)d_in[0];
  const float* Wq = (const float*)d_in[1];
  const float* Wk = (const float*)d_in[2];
  const float* Wv = (const float*)d_in[3];
  const float* Wo = (const float*)d_in[4];
  const float* g1 = (const float*)d_in[5];
  const float* b1 = (const float*)d_in[6];
  const float* g2 = (const float*)d_in[7];
  const float* b2 = (const float*)d_in[8];
  const float* W1 = (const float*)d_in[9];
  const float* c1 = (const float*)d_in[10];
  const float* W2 = (const float*)d_in[11];
  const float* c2 = (const float*)d_in[12];
  float* outp = (float*)d_out;

  // workspace carve-up (256B aligned)
  size_t off = 0;
  auto carve = [&](size_t bytes) -> void* {
    void* p = (char*)d_ws + off;
    off += (bytes + 255) & ~(size_t)255;
    return p;
  };
  unsigned short* xb   = (unsigned short*)carve((size_t)R * E * 2);
  unsigned short* WqT  = (unsigned short*)carve((size_t)HE * E * 2);
  unsigned short* WkT  = (unsigned short*)carve((size_t)HE * E * 2);
  unsigned short* WvT  = (unsigned short*)carve((size_t)HE * E * 2);
  unsigned short* WoT  = (unsigned short*)carve((size_t)E * HE * 2);
  unsigned short* W1T  = (unsigned short*)carve((size_t)F1 * E * 2);
  unsigned short* W2T  = (unsigned short*)carve((size_t)E * F1 * 2);
  unsigned short* qb   = (unsigned short*)carve((size_t)R * HE * 2);
  unsigned short* kb   = (unsigned short*)carve((size_t)R * HE * 2);
  unsigned short* vTb  = (unsigned short*)carve((size_t)R * HE * 2);
  unsigned short* attn = (unsigned short*)carve((size_t)R * HE * 2);
  float*          ybuf = (float*)carve((size_t)R * E * 4);
  float*          nrmf = (float*)carve((size_t)R * E * 4);
  unsigned short* nrmb = (unsigned short*)carve((size_t)R * E * 2);
  unsigned short* hbuf = (unsigned short*)carve((size_t)R * F1 * 2);

  // 1) convert x and weights to bf16 (weights transposed to [N][K])
  f32_to_bf16_kernel<<<(R * E + 255) / 256, 256, 0, stream>>>(x, xb, R * E);
  transpose_to_bf16_kernel<<<dim3(HE / 32, E / 32), 256, 0, stream>>>(Wq, WqT, E, HE);
  transpose_to_bf16_kernel<<<dim3(HE / 32, E / 32), 256, 0, stream>>>(Wk, WkT, E, HE);
  transpose_to_bf16_kernel<<<dim3(HE / 32, E / 32), 256, 0, stream>>>(Wv, WvT, E, HE);
  transpose_to_bf16_kernel<<<dim3(E / 32, HE / 32), 256, 0, stream>>>(Wo, WoT, HE, E);
  transpose_to_bf16_kernel<<<dim3(F1 / 32, E / 32), 256, 0, stream>>>(W1, W1T, E, F1);
  transpose_to_bf16_kernel<<<dim3(E / 32, F1 / 32), 256, 0, stream>>>(W2, W2T, F1, E);

  // 2) Q/K/V projections (V stored transposed [B,HE,S])
  gemm_bf16_kernel<<<dim3(HE / 256, R / 64), 256, 0, stream>>>(
      xb, WqT, nullptr, nullptr, qb, nullptr, R, HE, E, /*mode=*/0, 0, S, HE);
  gemm_bf16_kernel<<<dim3(HE / 256, R / 64), 256, 0, stream>>>(
      xb, WkT, nullptr, nullptr, kb, nullptr, R, HE, E, /*mode=*/0, 0, S, HE);
  gemm_bf16_kernel<<<dim3(HE / 256, R / 64), 256, 0, stream>>>(
      xb, WvT, nullptr, nullptr, vTb, nullptr, R, HE, E, /*mode=*/2, 0, S, HE);

  // 3) flash attention (scale = 1/sqrt(E) per the reference)
  float scale = 1.0f / sqrtf((float)E);
  flash_attn_kernel<<<Bb * H * (S / 16), 256, 0, stream>>>(qb, kb, vTb, attn, S, E, H, scale);

  // 4) output projection + residual -> ybuf (fp32), then LayerNorm 1
  gemm_bf16_kernel<<<dim3(E / 256, R / 64), 256, 0, stream>>>(
      attn, WoT, nullptr, x, nullptr, ybuf, R, E, HE, /*mode=*/1, 0, S, HE);
  layernorm_kernel<<<R, 256, 0, stream>>>(ybuf, g1, b1, nrmf, nrmb, E);

  // 5) FFN: relu(normed @ W1 + c1) -> hbuf (bf16)
  gemm_bf16_kernel<<<dim3(F1 / 256, R / 64), 256, 0, stream>>>(
      nrmb, W1T, c1, nullptr, hbuf, nullptr, R, F1, E, /*mode=*/0, /*relu=*/1, S, HE);

  // 6) h @ W2 + c2 + normed -> ybuf (fp32), then LayerNorm 2 into d_out
  gemm_bf16_kernel<<<dim3(E / 256, R / 64), 256, 0, stream>>>(
      hbuf, W2T, c2, nrmf, nullptr, ybuf, R, E, F1, /*mode=*/1, 0, S, HE);
  layernorm_kernel<<<R, 256, 0, stream>>>(ybuf, g2, b2, outp, nullptr, E);
}
